// PC_RNN_GC_HC_A_14413910245620
// MI455X (gfx1250) — matrix operand order repaired and checked by reading, not compiled, not run on hardware
//
#include <hip/hip_runtime.h>

// ---------------------------------------------------------------------------
// Persistent predictive-coding RNN scan for gfx1250 (MI455X).
// 32 workgroups x 256 threads (8 waves, wave32). Each WG owns 16 batch rows
// and iterates the full 1024-step scan. Weights live in LDS as f16 (plus
// transposed copies so every WMMA B-tile is a contiguous ds_load_b128 pair).
// h / hp / tanh state lives in registers (same wave owns the same N-tiles in
// every phase). Matmuls use v_wmma_f32_16x16x32_f16; tanh uses the gfx1250
// hardware v_tanh_f32 (TRANS32, co-executes with the XDL/WMMA pipe).
// x(t) is register-prefetched before the step's first barrier; x(t+1) is
// L2-prefetched (global_prefetch_b8) by the waves idle in Phase B.
// ---------------------------------------------------------------------------

typedef _Float16 h16;
typedef __attribute__((ext_vector_type(8)))  _Float16 v8h;
typedef __attribute__((ext_vector_type(16))) _Float16 v16h;
typedef __attribute__((ext_vector_type(8)))  float    v8f;

#define WMMA_F16(a, b, c) \
  __builtin_amdgcn_wmma_f32_16x16x32_f16(false, (a), false, (b), (short)0, (c), false, false)

// gfx1250 hardware tanh (TRANS32) when the toolchain exposes it.
#if __has_builtin(__builtin_amdgcn_tanhf)
  #define FAST_TANH(x) __builtin_amdgcn_tanhf(x)
#elif __has_builtin(__builtin_amdgcn_tanh_f32)
  #define FAST_TANH(x) __builtin_amdgcn_tanh_f32(x)
#else
  #define FAST_TANH(x) tanhf(x)
#endif

namespace {

constexpr int SEQ    = 1024;
constexpr int BATCH  = 512;
constexpr int STATES = 256;
constexpr int OUTD   = 64;
constexpr int CAUSES = 32;
constexpr int BWG    = 16;   // batch rows per workgroup
constexpr int NWG    = BATCH / BWG;   // 32 workgroups

constexpr float INV_TAU  = 0.1f;      // 1/10
constexpr float ALPHA_X  = 0.1f;
constexpr float ALPHA_H  = 0.1f;
constexpr float ALPHA_HP = 0.001f;

// ---- LDS carve (bytes) ----------------------------------------------------
constexpr int OFF_WR   = 0;                       // [256][256] f16 (w_r row-major)
constexpr int OFF_WO   = OFF_WR   + 256*256*2;    // [64][256]  f16 (w_o row-major)
constexpr int OFF_WOT  = OFF_WO   + 64*256*2;     // [256][64]  f16 (w_o^T)
constexpr int OFF_WC   = OFF_WOT  + 256*64*2;     // [256][32]  f16 (w_c row-major)
constexpr int OFF_WCT  = OFF_WC   + 256*32*2;     // [32][256]  f16 (w_c^T)
constexpr int OFF_THA  = OFF_WCT  + 32*256*2;     // [16][256] f16  tanh(h_post)
constexpr int OFF_THB  = OFF_THA  + 16*256*2;     // [16][256] f16  tanh(h_prior)
constexpr int OFF_ERR  = OFF_THB  + 16*256*2;     // [16][64]  f16  error
constexpr int OFF_EHP  = OFF_ERR  + 16*64*2;      // [16][256] f16  error_hp
constexpr int OFF_CA   = OFF_EHP  + 16*256*2;     // [16][32]  f16  causes
constexpr int SMEM_BYTES = OFF_CA + 16*32*2;      // ~252 KB (< 320 KB WGP LDS)

// A-operand tile load: A is row-major [16][stride] f16 in LDS, base already
// offset to (row 0, k0). Lanes 0-15 hold row=lane, K {0..7,16..23};
// lanes 16-31 hold row=lane-16, K {8..15,24..31}.  (ISA 7.12.2, 16-bit A)
__device__ __forceinline__ v16h ld_a(const h16* base, int stride, int lane) {
  const int r  = lane & 15;
  const int kb = (lane >> 4) << 3;          // 0 or 8
  const h16* p = base + r * stride + kb;
  v8h lo = *(const v8h*)(p);                // K = kb .. kb+7
  v8h hi = *(const v8h*)(p + 16);           // K = kb+16 .. kb+23
  return __builtin_shufflevector(lo, hi, 0,1,2,3,4,5,6,7,8,9,10,11,12,13,14,15);
}

// B-operand tile load: B(k,n) = W[n0+n][k0+k] where W is a row-major [N][stride]
// f16 LDS image (i.e. pre-transposed weight). base already offset to k0.
// Lanes 0-15: col=lane, K {0..7,16..23}; lanes 16-31: col=lane-16, K {8..15,24..31}.
__device__ __forceinline__ v16h ld_b(const h16* base, int stride, int n0, int lane) {
  const int col = lane & 15;
  const int kb  = (lane >> 4) << 3;         // 0 or 8
  const h16* p  = base + (n0 + col) * stride + kb;
  v8h lo = *(const v8h*)(p);
  v8h hi = *(const v8h*)(p + 16);
  return __builtin_shufflevector(lo, hi, 0,1,2,3,4,5,6,7,8,9,10,11,12,13,14,15);
}

} // namespace

extern "C" __global__ __launch_bounds__(256, 1)
void pcrnn_scan_kernel(const float* __restrict__ x,       // [SEQ][BATCH][OUTD]
                       const float* __restrict__ c_init,  // [BATCH][CAUSES]
                       const float* __restrict__ h_init,  // [BATCH][STATES]
                       const float* __restrict__ w_o,     // [OUTD][STATES]
                       const float* __restrict__ b_o,     // [OUTD]
                       const float* __restrict__ w_c,     // [STATES][CAUSES]
                       const float* __restrict__ w_r,     // [STATES][STATES]
                       const float* __restrict__ b_r,     // [STATES]
                       float* __restrict__ out)           // [SEQ][BATCH][OUTD]
{
  extern __shared__ char smem[];
  h16* s_wr  = (h16*)(smem + OFF_WR);
  h16* s_wo  = (h16*)(smem + OFF_WO);
  h16* s_wot = (h16*)(smem + OFF_WOT);
  h16* s_wc  = (h16*)(smem + OFF_WC);
  h16* s_wct = (h16*)(smem + OFF_WCT);
  h16* s_thA = (h16*)(smem + OFF_THA);
  h16* s_thB = (h16*)(smem + OFF_THB);
  h16* s_err = (h16*)(smem + OFF_ERR);
  h16* s_ehp = (h16*)(smem + OFF_EHP);
  h16* s_cA  = (h16*)(smem + OFF_CA);

  const int tid  = threadIdx.x;
  const int lane = tid & 31;
  const int wid  = tid >> 5;            // 8 waves
  const int b0   = blockIdx.x * BWG;    // first batch row of this WG
  const int colL = lane & 15;                     // N within a 16x16 tile
  const int rowL = ((lane >> 4) << 3);            // +0 / +8 row offset for C/D regs

  // ---- one-time weight conversion f32 -> f16 (+ transposed copies) --------
  for (int i = tid; i < STATES * STATES; i += 256) s_wr[i] = (h16)w_r[i];
  for (int i = tid; i < OUTD * STATES; i += 256) {
    const int o = i / STATES, s = i - o * STATES;
    const float v = w_o[i];
    s_wo[i]             = (h16)v;    // row-major [o][s]  (B for th @ w_o^T)
    s_wot[s * OUTD + o] = (h16)v;    // [s][o]            (B for err @ w_o)
  }
  for (int i = tid; i < STATES * CAUSES; i += 256) {
    const int s = i / CAUSES, c = i - s * CAUSES;
    const float v = w_c[i];
    s_wc[i]               = (h16)v;  // row-major [s][c]  (B for c @ w_c^T)
    s_wct[c * STATES + s] = (h16)v;  // [c][s]            (B for ehp @ w_c)
  }

  // tanh(h_init) as the first A-operand image
  for (int i = tid; i < BWG * STATES; i += 256) {
    const int r = i >> 8, s = i & 255;
    s_thA[i] = (h16)FAST_TANH(h_init[(size_t)(b0 + r) * STATES + s]);
  }

  // ---- register-resident state: this wave owns N-tiles 2*wid, 2*wid+1 -----
  float hA[2][8], hpA[2][8], thv[2][8], creg[8], xreg[8];
  float br_c[2];     // b_r[column] for this lane's two owned tiles (loop-inv)
  #pragma unroll
  for (int j = 0; j < 2; ++j) {
    const int nt = wid * 2 + j;
    br_c[j] = b_r[nt * 16 + colL];
    #pragma unroll
    for (int v = 0; v < 8; ++v) {
      hA[j][v]  = h_init[(size_t)(b0 + v + rowL) * STATES + nt * 16 + colL];
      hpA[j][v] = 0.0f;
    }
  }
  const float bo_c = (wid < 4) ? b_o[wid * 16 + colL] : 0.0f;  // loop-inv
  if (wid < 2) {   // waves 0-1 own the 16x32 causes block
    #pragma unroll
    for (int v = 0; v < 8; ++v) {
      creg[v] = c_init[(size_t)(b0 + v + rowL) * CAUSES + wid * 16 + colL];
      s_cA[(v + rowL) * CAUSES + wid * 16 + colL] = (h16)creg[v];
    }
  }

  // ======================= sequential scan =================================
  for (int t = 0; t < SEQ; ++t) {
    // x(t) depends on nothing: start its global loads before the barrier so
    // HBM latency hides under the Phase-A WMMA chain. Idle-in-B waves warm
    // L2 for x(t+1) via global_prefetch_b8.
    if (wid < 4) {
      #pragma unroll
      for (int v = 0; v < 8; ++v) {
        const size_t gi =
            ((size_t)t * BATCH + (b0 + v + rowL)) * OUTD + wid * 16 + colL;
        xreg[v] = x[gi];
      }
    } else if (t + 1 < SEQ) {
      const size_t gp = ((size_t)(t + 1) * BATCH + (b0 + (lane & 15))) * OUTD
                      + ((wid - 4) & 1) * 32 + (lane >> 4) * 16;
      __builtin_prefetch(&x[gp], 0, 1);
    }

    __syncthreads();   // thA / cA (from prev step) visible

    // ---- Phase A: hp_prior, h_prior, tanh(h_prior) ------------------------
    #pragma unroll
    for (int j = 0; j < 2; ++j) {
      const int nt = wid * 2 + j;
      v8f acc = {};
      #pragma unroll
      for (int k0 = 0; k0 < STATES; k0 += 32) {
        v16h a = ld_a(s_thA + k0, STATES, lane);
        v16h b = ld_b(s_wr + k0, STATES, nt * 16, lane);
        acc = WMMA_F16(a, b, acc);
      }
      { // + c @ w_c^T  (K = 32 exactly)
        v16h a = ld_a(s_cA, CAUSES, lane);
        v16h b = ld_b(s_wc, CAUSES, nt * 16, lane);
        acc = WMMA_F16(a, b, acc);
      }
      #pragma unroll
      for (int v = 0; v < 8; ++v) {
        const int cg = nt * 16 + colL;
        const float hp_pr = (1.0f - INV_TAU) * hpA[j][v]
                          + INV_TAU * (acc[v] + br_c[j] - hA[j][v]);
        const float h_pr = hA[j][v] + hp_pr;
        const float th   = FAST_TANH(h_pr);
        hA[j][v]  = h_pr;      // h := h_prior (h_post applied in Phase C)
        hpA[j][v] = hp_pr;
        thv[j][v] = th;
        s_thB[(v + rowL) * STATES + cg] = (h16)th;
      }
    }
    __syncthreads();   // thB ready

    // ---- Phase B: x_pred, error (waves 0-3, one 16x16 OUT tile each) ------
    if (wid < 4) {
      v8f acc = {};
      #pragma unroll
      for (int k0 = 0; k0 < STATES; k0 += 32) {
        v16h a = ld_a(s_thB + k0, STATES, lane);
        v16h b = ld_b(s_wo + k0, STATES, wid * 16, lane);
        acc = WMMA_F16(a, b, acc);
      }
      #pragma unroll
      for (int v = 0; v < 8; ++v) {
        const int og = wid * 16 + colL;
        const size_t gi = ((size_t)t * BATCH + (b0 + v + rowL)) * OUTD + og;
        const float e = acc[v] + bo_c - xreg[v];
        s_err[(v + rowL) * OUTD + og] = (h16)e;
        out[gi] = e;                               // scan output
      }
    }
    __syncthreads();   // errA ready

    // ---- Phase C: h_post, hp_post, error_hp, tanh(h_post) -----------------
    #pragma unroll
    for (int j = 0; j < 2; ++j) {
      const int nt = wid * 2 + j;
      v8f acc = {};
      #pragma unroll
      for (int k0 = 0; k0 < OUTD; k0 += 32) {        // err @ w_o, K = 64
        v16h a = ld_a(s_err + k0, OUTD, lane);
        v16h b = ld_b(s_wot + k0, OUTD, nt * 16, lane);
        acc = WMMA_F16(a, b, acc);
      }
      #pragma unroll
      for (int v = 0; v < 8; ++v) {
        const int cg = nt * 16 + colL;
        const float th = thv[j][v];
        const float eh = ALPHA_X * (1.0f - th * th) * acc[v];
        hA[j][v]  -= eh;                 // h_post
        hpA[j][v] -= ALPHA_H * eh;       // hp_post
        s_ehp[(v + rowL) * STATES + cg] = (h16)(ALPHA_H * eh);
        s_thA[(v + rowL) * STATES + cg] = (h16)FAST_TANH(hA[j][v]); // next step
      }
    }
    __syncthreads();   // ehpA ready

    // ---- Phase D: causes update (waves 0-1) -------------------------------
    if (wid < 2) {
      v8f acc = {};
      #pragma unroll
      for (int k0 = 0; k0 < STATES; k0 += 32) {      // ehp @ w_c, K = 256
        v16h a = ld_a(s_ehp + k0, STATES, lane);
        v16h b = ld_b(s_wct + k0, STATES, wid * 16, lane);
        acc = WMMA_F16(a, b, acc);
      }
      #pragma unroll
      for (int v = 0; v < 8; ++v) {
        creg[v] -= ALPHA_HP * acc[v];
        s_cA[(v + rowL) * CAUSES + wid * 16 + colL] = (h16)creg[v];
      }
    }
  }
}

extern "C" void kernel_launch(void* const* d_in, const int* in_sizes, int n_in,
                              void* d_out, int out_size, void* d_ws, size_t ws_size,
                              hipStream_t stream) {
  (void)in_sizes; (void)n_in; (void)d_ws; (void)ws_size; (void)out_size;
  const float* x      = (const float*)d_in[0];
  const float* c_init = (const float*)d_in[1];
  const float* h_init = (const float*)d_in[2];
  const float* w_o    = (const float*)d_in[3];
  const float* b_o    = (const float*)d_in[4];
  const float* w_c    = (const float*)d_in[5];
  const float* w_r    = (const float*)d_in[6];
  const float* b_r    = (const float*)d_in[7];
  float* out = (float*)d_out;

  // ~252 KB dynamic LDS per workgroup (WGP has 320 KB) — raise the cap.
  hipFuncSetAttribute((const void*)pcrnn_scan_kernel,
                      hipFuncAttributeMaxDynamicSharedMemorySize, SMEM_BYTES);

  pcrnn_scan_kernel<<<NWG, 256, SMEM_BYTES, stream>>>(
      x, c_init, h_init, w_o, b_o, w_c, w_r, b_r, out);
}
